// SelfAttention_62654982914287
// MI455X (gfx1250) — matrix operand order repaired and checked by reading, not compiled, not run on hardware
//
#include <hip/hip_runtime.h>
#include <hip/hip_bf16.h>

typedef _Float16 half_t;
typedef __attribute__((ext_vector_type(8)))  _Float16 v8h;
typedef __attribute__((ext_vector_type(16))) _Float16 v16h;
typedef __attribute__((ext_vector_type(8)))  float    v8f;

#define QN 32768
#define QD 1024
#define QD4 256

// ---------------------------------------------------------------------------
// Weight expansion: qlinear(x, w) == x @ W_eff where W_eff is the 1024x1024
// quaternion block matrix. Stored TRANSPOSED: wt[n][k] (K contiguous) which is
// the WMMA B-fragment-friendly order.
// ---------------------------------------------------------------------------
__global__ void prep_w_kernel(const float* __restrict__ w, half_t* __restrict__ wt,
                              float scale) {
    int idx = blockIdx.x * blockDim.x + threadIdx.x;   // 0 .. 1024*1024
    if (idx >= QD * QD) return;
    int n = idx >> 10;          // output column
    int k = idx & 1023;         // input row
    int nb = n >> 8;            // output quaternion component block
    int kb = k >> 8;            // input quaternion component block
    const int   cidx[4][4] = {{0,1,2,3},{1,0,3,2},{2,3,0,1},{3,2,1,0}};
    const float sgn [4][4] = {{ 1.f, 1.f, 1.f, 1.f},
                              {-1.f, 1.f, 1.f,-1.f},
                              {-1.f,-1.f, 1.f, 1.f},
                              {-1.f, 1.f,-1.f, 1.f}};
    int   c = cidx[kb][nb];
    float s = sgn[kb][nb] * scale;
    float val = w[((c << 8) + (k & 255)) * QD4 + (n & 255)];  // w[c][k%256][n%256]
    wt[idx] = (half_t)(s * val);
}

__global__ void convert_x_kernel(const float* __restrict__ x, half_t* __restrict__ xh) {
    int idx = (blockIdx.x * blockDim.x + threadIdx.x) * 4;
    if (idx >= QN * QD) return;
    #pragma unroll
    for (int t = 0; t < 4; ++t) xh[idx + t] = (half_t)x[idx + t];
}

// ---------------------------------------------------------------------------
// Fragment loaders per CDNA5 ISA 7.12.2 (wave32, 16-bit operands):
//   A lane: row M=lane%16, halves[0..7] at K=k0+(lane/16)*8, halves[8..15] +16.
//   B lane: col N=lane%16, halves[0..7] at K=k0+(lane/16)*8, halves[8..15] +16.
// pa/pb are per-lane base pointers; all pieces are immediate offsets.
// ---------------------------------------------------------------------------
template <int K>
__device__ __forceinline__ void load_a4(v16h a[4], const half_t* __restrict__ pa) {
    #pragma unroll
    for (int i = 0; i < 4; ++i) {
        v8h lo = *(const v8h*)(pa + i * 16 * K);
        v8h hi = *(const v8h*)(pa + i * 16 * K + 16);
        a[i] = __builtin_shufflevector(lo, hi, 0,1,2,3,4,5,6,7,8,9,10,11,12,13,14,15);
    }
}

template <int K>
__device__ __forceinline__ void load_b2(v16h b[2], const half_t* __restrict__ pb) {
    #pragma unroll
    for (int j = 0; j < 2; ++j) {
        v8h lo = *(const v8h*)(pb + j * 16 * K);
        v8h hi = *(const v8h*)(pb + j * 16 * K + 16);
        b[j] = __builtin_shufflevector(lo, hi, 0,1,2,3,4,5,6,7,8,9,10,11,12,13,14,15);
    }
}

__device__ __forceinline__ void mma8(v8f acc[4][2], const v16h a[4], const v16h b[2]) {
    #pragma unroll
    for (int i = 0; i < 4; ++i)
        #pragma unroll
        for (int j = 0; j < 2; ++j)
            acc[i][j] = __builtin_amdgcn_wmma_f32_16x16x32_f16(
                false, a[i], false, b[j], (short)0, acc[i][j], false, false);
}

// ---------------------------------------------------------------------------
// GEMM: C[M x Ncols] = A[M x K] * Bt^T   (Bt is [Ncols x K], K contiguous)
// Block: 256 threads = 8 waves in 2(M) x 4(N) -> 128x128 block tile.
// Wave: 64x32 tile = 4x2 WMMA tiles; K pipelined 2 x 32 deep (double buffer).
// __launch_bounds__(256, 1): allow full register budget (no spills) since the
// pipelined schedule hides latency within a wave.
// ---------------------------------------------------------------------------
template <int K, bool STORE_F16>
__device__ __forceinline__ void gemm_wmma_body(const half_t* __restrict__ A,
                                               const half_t* __restrict__ Bt,
                                               void* __restrict__ C, int ldc) {
    const int lane = threadIdx.x & 31;
    const int wave = threadIdx.x >> 5;          // 0..7
    const int m0 = blockIdx.y * 128 + (wave >> 2) * 64;   // 2 waves in M
    const int n0 = blockIdx.x * 128 + (wave & 3) * 32;    // 4 waves in N

    const half_t* pa = A  + (size_t)(m0 + (lane & 15)) * K + ((lane >> 4) << 3);
    const half_t* pb = Bt + (size_t)(n0 + (lane & 15)) * K + ((lane >> 4) << 3);

    v8f  acc[4][2] = {};
    v16h a0[4], b0[2], a1[4], b1[2];

    // prologue: buffer 0 for kk=0
    load_a4<K>(a0, pa);
    load_b2<K>(b0, pb);

    for (int kk = 0; kk < K; kk += 64) {
        // buffer 1 for kk+32 (K % 64 == 0 so always in range)
        load_a4<K>(a1, pa + 32);
        load_b2<K>(b1, pb + 32);
        mma8(acc, a0, b0);
        if (kk + 64 < K) {              // buffer 0 for kk+64
            load_a4<K>(a0, pa + 64);
            load_b2<K>(b0, pb + 64);
        }
        mma8(acc, a1, b1);
        pa += 64;
        pb += 64;
    }

    // C layout: elem r -> row (lane/16)*8 + r, col lane%16
    const int rbase = (lane >> 4) << 3;
    const int cofs  = lane & 15;
    #pragma unroll
    for (int i = 0; i < 4; ++i) {
        #pragma unroll
        for (int j = 0; j < 2; ++j) {
            #pragma unroll
            for (int r = 0; r < 8; ++r) {
                size_t row = (size_t)(m0 + i * 16 + rbase + r);
                size_t col = (size_t)(n0 + j * 16 + cofs);
                if (STORE_F16) ((half_t*)C)[row * ldc + col] = (half_t)acc[i][j][r];
                else           ((float*) C)[row * ldc + col] = acc[i][j][r];
            }
        }
    }
}

__global__ void __launch_bounds__(256, 1)
gemm_wmma_f16out(const half_t* __restrict__ A, const half_t* __restrict__ Bt,
                 half_t* __restrict__ C, int ldc) {
    gemm_wmma_body<QD, true>(A, Bt, (void*)C, ldc);
}

__global__ void __launch_bounds__(256, 1)
gemm_wmma_f32out(const half_t* __restrict__ A, const half_t* __restrict__ Bt,
                 float* __restrict__ C, int ldc) {
    gemm_wmma_body<QD, false>(A, Bt, (void*)C, ldc);
}

// ---------------------------------------------------------------------------
// Elementwise: logits = hamilton(Q,K); att = q_normalize(logits) -> d_out[N*D..)
//              y1 = hamilton(att, V) -> f16 scratch for the output qlinear
// ---------------------------------------------------------------------------
__global__ void qattn_elem_kernel(const half_t* __restrict__ qkv,
                                  float* __restrict__ att_out,
                                  half_t* __restrict__ y1h) {
    int idx = blockIdx.x * blockDim.x + threadIdx.x;  // 0 .. N*256
    if (idx >= QN * QD4) return;
    int row = idx >> 8;
    int p   = idx & 255;
    const half_t* b = qkv + (size_t)row * 3072;

    float qr = (float)b[p],        qi = (float)b[p + 256];
    float qj = (float)b[p + 512],  qk = (float)b[p + 768];
    float kr = (float)b[1024 + p],       ki = (float)b[1024 + p + 256];
    float kj = (float)b[1024 + p + 512], kk = (float)b[1024 + p + 768];
    float vr = (float)b[2048 + p],       vi = (float)b[2048 + p + 256];
    float vj = (float)b[2048 + p + 512], vk = (float)b[2048 + p + 768];

    float lr = qr * kr - qi * ki - qj * kj - qk * kk;
    float li = qr * ki + qi * kr + qj * kk - qk * kj;
    float lj = qr * kj - qi * kk + qj * kr + qk * ki;
    float lk = qr * kk + qi * kj - qj * ki + qk * kr;

    float inv = 1.0f / __builtin_sqrtf(lr * lr + li * li + lj * lj + lk * lk + 1e-4f);
    float ar = lr * inv, ai = li * inv, aj = lj * inv, ak = lk * inv;

    float* ao = att_out + (size_t)row * QD;
    ao[p] = ar; ao[p + 256] = ai; ao[p + 512] = aj; ao[p + 768] = ak;

    float yr = ar * vr - ai * vi - aj * vj - ak * vk;
    float yi = ar * vi + ai * vr + aj * vk - ak * vj;
    float yj = ar * vj - ai * vk + aj * vr + ak * vi;
    float yk = ar * vk + ai * vj - aj * vi + ak * vr;

    half_t* yo = y1h + (size_t)row * QD;
    yo[p] = (half_t)yr; yo[p + 256] = (half_t)yi;
    yo[p + 512] = (half_t)yj; yo[p + 768] = (half_t)yk;
}

// ---------------------------------------------------------------------------
extern "C" void kernel_launch(void* const* d_in, const int* in_sizes, int n_in,
                              void* d_out, int out_size, void* d_ws, size_t ws_size,
                              hipStream_t stream) {
    const float* x  = (const float*)d_in[0];
    const float* wq = (const float*)d_in[1];
    const float* wk = (const float*)d_in[2];
    const float* wv = (const float*)d_in[3];
    const float* wo = (const float*)d_in[4];

    float* y_out   = (float*)d_out;                    // [N x 1024]
    float* att_out = (float*)d_out + (size_t)QN * QD;  // [N x 1024]

    // Workspace: xh 64MB | wt_qkv 6MB | wt_o 2MB | qkvh 192MB | y1h 64MB
    half_t* xh     = (half_t*)d_ws;
    half_t* wt_qkv = xh     + (size_t)QN * QD;
    half_t* wt_o   = wt_qkv + (size_t)3 * QD * QD;
    half_t* qkvh   = wt_o   + (size_t)QD * QD;
    half_t* y1h    = qkvh   + (size_t)QN * 3072;

    // 1) expand + transpose weights to f16 (0.06 Q-scale folded into Wq)
    {
        dim3 blk(256), grd((QD * QD + 255) / 256);
        prep_w_kernel<<<grd, blk, 0, stream>>>(wq, wt_qkv,                   0.06f);
        prep_w_kernel<<<grd, blk, 0, stream>>>(wk, wt_qkv + (size_t)QD * QD, 1.0f);
        prep_w_kernel<<<grd, blk, 0, stream>>>(wv, wt_qkv + (size_t)2*QD*QD, 1.0f);
        prep_w_kernel<<<grd, blk, 0, stream>>>(wo, wt_o,                     1.0f);
    }

    // 2) x -> f16
    convert_x_kernel<<<(QN * QD / 4 + 255) / 256, 256, 0, stream>>>(x, xh);

    // 3) fused QKV GEMM: [32768 x 1024] x [1024 x 3072] -> qkvh f16
    {
        dim3 blk(256), grd(3072 / 128, QN / 128);
        gemm_wmma_f16out<<<grd, blk, 0, stream>>>(xh, wt_qkv, qkvh, 3072);
    }

    // 4) hamilton / normalize / hamilton  (att -> d_out, y1 -> scratch)
    qattn_elem_kernel<<<(QN * QD4 + 255) / 256, 256, 0, stream>>>(qkvh, att_out, y1h);

    // 5) output qlinear: [32768 x 1024] x [1024 x 1024] -> y (f32, d_out)
    {
        dim3 blk(256), grd(QD / 128, QN / 128);
        gemm_wmma_f32out<<<grd, blk, 0, stream>>>(y1h, wt_o, y_out, QD);
    }
}
